// MambaBlock_73512660238844
// MI455X (gfx1250) — compile-verified, matrix operand order checked by reading, not run on hardware
//
#include <hip/hip_runtime.h>
#include <cstdint>

#define BSZ     2
#define SEQL    2048
#define DMODEL  1024
#define DINNER  2048
#define DSTATE  16
#define NGROUPS 4
#define NTOK    (BSZ * SEQL)
#define EPSF    1e-5f

typedef __attribute__((ext_vector_type(16))) __bf16 v16bf;
typedef __attribute__((ext_vector_type(8)))  float  v8f;

// ---------------------------------------------------------------------------
// WMMA GEMM: C[m,n] = sum_k A[m,k] * W[n,k]  (act @ W^T), bf16 in / f32 acc.
// Block = 128 threads = 4 waves (2x2), block tile 64x64, K-step 64.
// A/B 64x64 bf16 tiles are staged in LDS with global_load_async_to_lds_b128
// (ASYNCcnt), double buffered; fragments are read from LDS (ds_load_b128).
// Fragment scheme: lane L holds 16 contiguous-K bf16 of row (L%16) at
// k + (L>>4)*16 -- identical placement for A and B fragments, so the
// K-permutation cancels in the WMMA dot product.
// EPI: 0 gated=clip(acc+b,-1,1)->bf16 | 1 x+acc+b ->f32 + group absmax
//      2 plain f32 | 3 f32+bf16 | 4 softplus(acc+b) f32 | 5 resid+acc ->f32
// ---------------------------------------------------------------------------
template <int EPI>
__global__ __launch_bounds__(128) void gemm_bf16_kernel(
    const __bf16* __restrict__ A, int lda, const __bf16* __restrict__ Bw,
    int ldb, int K, float* __restrict__ Cout, __bf16* __restrict__ CoutBf,
    int ldc, const float* __restrict__ bias, const float* __restrict__ resid,
    unsigned* __restrict__ gammaU) {
  __shared__ __bf16 tA[2][64 * 64];
  __shared__ __bf16 tB[2][64 * 64];

  const int tid  = threadIdx.x;
  const int lane = tid & 31;
  const int wave = tid >> 5;
  const int wm = wave >> 1, wn = wave & 1;
  const int mBlk = blockIdx.x * 64;
  const int nBlk = blockIdx.y * 64;
  const int r    = lane & 15;
  const int half = lane >> 4;

  // stage one 64x64 bf16 tile of A and of B into LDS buffer `buf`
  auto issue = [&](int buf, int k0) {
    const uint32_t ldsA = (uint32_t)(uintptr_t)(&tA[buf][0]);
    const uint32_t ldsB = (uint32_t)(uintptr_t)(&tB[buf][0]);
#pragma unroll
    for (int s = 0; s < 4; ++s) {
      const int seg = tid + s * 128;        // 512 x 16B segments per tile
      const int row = seg >> 3;
      const int c16 = seg & 7;
      const __bf16* ga = A  + (size_t)(mBlk + row) * lda + k0 + c16 * 8;
      const __bf16* gb = Bw + (size_t)(nBlk + row) * ldb + k0 + c16 * 8;
      const uint32_t la = ldsA + (uint32_t)seg * 16u;
      const uint32_t lb = ldsB + (uint32_t)seg * 16u;
      asm volatile("global_load_async_to_lds_b128 %0, %1, off"
                   :: "v"(la), "v"(ga) : "memory");
      asm volatile("global_load_async_to_lds_b128 %0, %1, off"
                   :: "v"(lb), "v"(gb) : "memory");
      if (k0 + 128 <= K) __builtin_prefetch(ga + 64, 0, 1);  // tile kt+2
    }
  };

  v8f c00 = {}, c01 = {}, c10 = {}, c11 = {};
  const int NT = K >> 6;
  issue(0, 0);
  for (int kt = 0; kt < NT; ++kt) {
    const int cur = kt & 1;
    asm volatile("s_wait_asynccnt 0x0" ::: "memory");
    __syncthreads();  // all waves' async tiles for `cur` are resident
    if (kt + 1 < NT) issue(cur ^ 1, (kt + 1) << 6);
#pragma unroll
    for (int kk = 0; kk < 64; kk += 32) {
      const __bf16* pa = &tA[cur][(wm * 32 + r) * 64 + kk + half * 16];
      const __bf16* pb = &tB[cur][(wn * 32 + r) * 64 + kk + half * 16];
      v16bf a0 = *(const v16bf*)pa;
      v16bf a1 = *(const v16bf*)(pa + 16 * 64);
      v16bf b0 = *(const v16bf*)pb;
      v16bf b1 = *(const v16bf*)(pb + 16 * 64);
      c00 = __builtin_amdgcn_wmma_f32_16x16x32_bf16(false, a0, false, b0,
                                                    (short)0, c00, false, false);
      c01 = __builtin_amdgcn_wmma_f32_16x16x32_bf16(false, a0, false, b1,
                                                    (short)0, c01, false, false);
      c10 = __builtin_amdgcn_wmma_f32_16x16x32_bf16(false, a1, false, b0,
                                                    (short)0, c10, false, false);
      c11 = __builtin_amdgcn_wmma_f32_16x16x32_bf16(false, a1, false, b1,
                                                    (short)0, c11, false, false);
    }
  }

  const int m0 = mBlk + wm * 32;
  const int n0 = nBlk + wn * 32;
  float lmax = 0.f;
  auto emit = [&](const v8f& c, int ti, int tj) {
    const int gm = m0 + ti * 16 + half * 8;  // C layout: vgpr v -> row v+8*half
    const int gn = n0 + tj * 16 + r;         // lane%16 -> column
#pragma unroll
    for (int v = 0; v < 8; ++v) {
      float val = c[v];
      const size_t idx = (size_t)(gm + v) * ldc + gn;
      if constexpr (EPI == 0) {
        val = fminf(fmaxf(val + bias[gn], -1.f), 1.f);
        CoutBf[idx] = (__bf16)val;
      } else if constexpr (EPI == 1) {
        val = val + bias[gn] + resid[idx];
        Cout[idx] = val;
        lmax = fmaxf(lmax, fabsf(val));
      } else if constexpr (EPI == 2) {
        Cout[idx] = val;
      } else if constexpr (EPI == 3) {
        Cout[idx] = val;
        CoutBf[idx] = (__bf16)val;
      } else if constexpr (EPI == 4) {
        val += bias[gn];
        val = (val > 20.f) ? val : log1pf(__expf(val));
        Cout[idx] = val;
      } else {
        Cout[idx] = val + resid[idx];
      }
    }
  };
  emit(c00, 0, 0);
  emit(c01, 0, 1);
  emit(c10, 1, 0);
  emit(c11, 1, 1);
  if constexpr (EPI == 1) {
    // 64-row block is fully inside one 512-token seq group (512 % 64 == 0)
    const int grp = ((blockIdx.x * 64) % SEQL) / (SEQL / NGROUPS);
    atomicMax(gammaU + grp, __float_as_uint(lmax));
  }
}

// --------------------------- LayerNorm -> bf16 -----------------------------
__global__ __launch_bounds__(256) void ln_bf16_kernel(
    const float* __restrict__ x, const float* __restrict__ g,
    const float* __restrict__ b, __bf16* __restrict__ out) {
  const int t = blockIdx.x;
  const float* xt = x + (size_t)t * DMODEL;
  float s = 0.f, s2 = 0.f;
  for (int i = threadIdx.x; i < DMODEL; i += 256) {
    float v = xt[i];
    s += v;
    s2 += v * v;
  }
#pragma unroll
  for (int o = 16; o; o >>= 1) {
    s += __shfl_xor(s, o, 32);
    s2 += __shfl_xor(s2, o, 32);
  }
  __shared__ float sb[16];
  const int wv = threadIdx.x >> 5;
  if ((threadIdx.x & 31) == 0) {
    sb[wv] = s;
    sb[8 + wv] = s2;
  }
  __syncthreads();
  if (threadIdx.x == 0) {
    float a = 0.f, c = 0.f;
    for (int i = 0; i < 8; ++i) {
      a += sb[i];
      c += sb[8 + i];
    }
    sb[0] = a;
    sb[8] = c;
  }
  __syncthreads();
  const float mean = sb[0] * (1.f / DMODEL);
  const float var = sb[8] * (1.f / DMODEL) - mean * mean;
  const float rstd = rsqrtf(var + EPSF);
  __bf16* ot = out + (size_t)t * DMODEL;
  for (int i = threadIdx.x; i < DMODEL; i += 256)
    ot[i] = (__bf16)((xt[i] - mean) * rstd * g[i] + b[i]);
}

// ------------------------ misc elementwise kernels -------------------------
__global__ void init_kernel(const float* __restrict__ A_log,
                            float* __restrict__ Aneg,
                            unsigned* __restrict__ gammaU) {
  const int i = blockIdx.x * blockDim.x + threadIdx.x;
  if (i < NGROUPS) gammaU[i] = 0u;
  if (i < DINNER * DSTATE) Aneg[i] = -__expf(A_log[i]);
}

__global__ void cvt_bf16_kernel(const float* __restrict__ in,
                                __bf16* __restrict__ out, int n, int signq) {
  const int i = blockIdx.x * blockDim.x + threadIdx.x;
  if (i >= n) return;
  float v = in[i];
  if (signq) v = (v > 0.f) ? 1.f : ((v < 0.f) ? -1.f : 0.f);
  out[i] = (__bf16)v;
}

__global__ void cvt_pad_xproj_kernel(const float* __restrict__ in,
                                     __bf16* __restrict__ out) {
  const int i = blockIdx.x * blockDim.x + threadIdx.x;
  if (i >= 128 * DINNER) return;
  const int row = i / DINNER, col = i % DINNER;
  out[i] = (__bf16)((row < 96) ? in[row * DINNER + col] : 0.f);
}

__global__ void quant_kernel(const float* __restrict__ x2,
                             const unsigned* __restrict__ gammaU,
                             float* __restrict__ qx) {
  const int i = blockIdx.x * blockDim.x + threadIdx.x;
  if (i >= NTOK * DMODEL) return;
  const int t = i / DMODEL;
  const int grp = (t % SEQL) / (SEQL / NGROUPS);
  const float gamma = __uint_as_float(gammaU[grp]);
  float v = x2[i] * (128.f / (gamma + EPSF));
  qx[i] = fminf(fmaxf(v, -128.f + EPSF), 128.f - EPSF);
}

// depthwise causal conv (D_CONV=4) + SiLU; xs = cols [0,2048) of xz
__global__ __launch_bounds__(256) void conv_silu_kernel(
    const float* __restrict__ xz, const float* __restrict__ cw,
    const float* __restrict__ cb, __bf16* __restrict__ xsc_bf) {
  const int t = blockIdx.x;
  const int b = t / SEQL, l = t % SEQL;
  for (int d = threadIdx.x; d < DINNER; d += 256) {
    float acc = cb[d];
#pragma unroll
    for (int j = 0; j < 4; ++j) {
      const int ll = l - 3 + j;
      if (ll >= 0)
        acc = fmaf(cw[d * 4 + j], xz[(size_t)(b * SEQL + ll) * 4096 + d], acc);
    }
    const float sv = acc / (1.f + __expf(-acc));
    xsc_bf[(size_t)t * DINNER + d] = (__bf16)sv;
  }
}

// selective scan: one channel per thread, 16-state recurrence in registers
__global__ __launch_bounds__(256) void scan_kernel(
    const float* __restrict__ dt, const __bf16* __restrict__ xsc_bf,
    const float* __restrict__ xz, const float* __restrict__ xdbl,
    const float* __restrict__ Aneg, const float* __restrict__ Dp,
    __bf16* __restrict__ ybf) {
  const int d = blockIdx.x * blockDim.x + threadIdx.x;  // 0..2047
  const int b = blockIdx.y;
  float h[DSTATE], Ar[DSTATE];
#pragma unroll
  for (int s = 0; s < DSTATE; ++s) {
    h[s] = 0.f;
    Ar[s] = Aneg[d * DSTATE + s];
  }
  const float Dd = Dp[d];
  for (int l = 0; l < SEQL; ++l) {
    const size_t t = (size_t)b * SEQL + l;
    const float dtv = dt[t * DINNER + d];
    const float xv = (float)xsc_bf[t * DINNER + d];
    const float zv = xz[t * 4096 + DINNER + d];
    const float* bc = xdbl + t * 128 + 64;  // B at [64,80), C at [80,96)
    const float p = dtv * xv;
    float y = 0.f;
#pragma unroll
    for (int s = 0; s < DSTATE; ++s) {
      const float dA = __expf(dtv * Ar[s]);
      h[s] = fmaf(h[s], dA, p * bc[s]);
      y = fmaf(h[s], bc[DSTATE + s], y);
    }
    const float yt = fmaf(Dd, xv, y);
    const float sg = zv / (1.f + __expf(-zv));
    ybf[t * DINNER + d] = (__bf16)(yt * sg);
  }
}

// ---------------------------------------------------------------------------
extern "C" void kernel_launch(void* const* d_in, const int* in_sizes, int n_in,
                              void* d_out, int out_size, void* d_ws,
                              size_t ws_size, hipStream_t stream) {
  (void)in_sizes; (void)n_in; (void)out_size; (void)ws_size;
  const float* x        = (const float*)d_in[0];
  const float* ln1_g    = (const float*)d_in[1];
  const float* ln1_b    = (const float*)d_in[2];
  const float* ln2_g    = (const float*)d_in[3];
  const float* ln2_b    = (const float*)d_in[4];
  const float* W1       = (const float*)d_in[5];
  const float* b1       = (const float*)d_in[6];
  const float* W2       = (const float*)d_in[7];
  const float* b2       = (const float*)d_in[8];
  const float* in_proj  = (const float*)d_in[9];
  const float* conv_w   = (const float*)d_in[10];
  const float* conv_b   = (const float*)d_in[11];
  const float* x_proj   = (const float*)d_in[12];
  const float* dt_projw = (const float*)d_in[13];
  const float* dt_projb = (const float*)d_in[14];
  const float* A_log    = (const float*)d_in[15];
  const float* D_param  = (const float*)d_in[16];
  const float* out_proj = (const float*)d_in[17];
  float* out = (float*)d_out;

  char* ws = (char*)d_ws;
  auto alloc = [&](size_t bytes) -> void* {
    void* p = (void*)ws;
    ws += (bytes + 255) & ~(size_t)255;
    return p;
  };
  __bf16*   qW1    = (__bf16*)alloc((size_t)DINNER * DMODEL * 2);
  __bf16*   qW2    = (__bf16*)alloc((size_t)DMODEL * DINNER * 2);
  __bf16*   wInBf  = (__bf16*)alloc((size_t)2 * DINNER * DMODEL * 2);
  __bf16*   wXpBf  = (__bf16*)alloc((size_t)128 * DINNER * 2);
  __bf16*   wDtBf  = (__bf16*)alloc((size_t)DINNER * 64 * 2);
  __bf16*   wOutBf = (__bf16*)alloc((size_t)DMODEL * DINNER * 2);
  float*    Aneg   = (float*)alloc((size_t)DINNER * DSTATE * 4);
  unsigned* gammaU = (unsigned*)alloc(256);
  __bf16*   lnBf   = (__bf16*)alloc((size_t)NTOK * DMODEL * 2);   // LN1 then LN2
  __bf16*   gatedBf= (__bf16*)alloc((size_t)NTOK * DINNER * 2);
  float*    x2     = (float*)alloc((size_t)NTOK * DMODEL * 4);
  float*    qx     = (float*)alloc((size_t)NTOK * DMODEL * 4);
  float*    xzBuf  = (float*)alloc((size_t)NTOK * 4096 * 4);
  __bf16*   xscBf  = (__bf16*)alloc((size_t)NTOK * DINNER * 2);
  float*    xdbl   = (float*)alloc((size_t)NTOK * 128 * 4);
  __bf16*   xdblBf = (__bf16*)alloc((size_t)NTOK * 128 * 2);
  float*    dtBuf  = (float*)alloc((size_t)NTOK * DINNER * 4);
  __bf16*   yBf    = (__bf16*)alloc((size_t)NTOK * DINNER * 2);

  const dim3 blk128(128), blk256(256);
  auto grids = [](int n) { return dim3((unsigned)((n + 255) / 256)); };

  // 0) init: gamma = 0, Aneg = -exp(A_log)
  init_kernel<<<grids(DINNER * DSTATE), blk256, 0, stream>>>(A_log, Aneg, gammaU);
  // 1) weight conversions
  cvt_bf16_kernel<<<grids(DINNER * DMODEL), blk256, 0, stream>>>(W1, qW1, DINNER * DMODEL, 1);
  cvt_bf16_kernel<<<grids(DMODEL * DINNER), blk256, 0, stream>>>(W2, qW2, DMODEL * DINNER, 1);
  cvt_bf16_kernel<<<grids(2 * DINNER * DMODEL), blk256, 0, stream>>>(in_proj, wInBf, 2 * DINNER * DMODEL, 0);
  cvt_pad_xproj_kernel<<<grids(128 * DINNER), blk256, 0, stream>>>(x_proj, wXpBf);
  cvt_bf16_kernel<<<grids(DINNER * 64), blk256, 0, stream>>>(dt_projw, wDtBf, DINNER * 64, 0);
  cvt_bf16_kernel<<<grids(DMODEL * DINNER), blk256, 0, stream>>>(out_proj, wOutBf, DMODEL * DINNER, 0);

  // 2) LN1(x) -> bf16
  ln_bf16_kernel<<<dim3(NTOK), blk256, 0, stream>>>(x, ln1_g, ln1_b, lnBf);
  // 3) gated = clip(LN1 @ sign(W1)^T + b1)
  gemm_bf16_kernel<0><<<dim3(NTOK / 64, DINNER / 64), blk128, 0, stream>>>(
      lnBf, DMODEL, qW1, DMODEL, DMODEL, nullptr, gatedBf, DINNER, b1, nullptr, nullptr);
  // 4) x2 = x + gated @ sign(W2)^T + b2, fused group abs-max
  gemm_bf16_kernel<1><<<dim3(NTOK / 64, DMODEL / 64), blk128, 0, stream>>>(
      gatedBf, DINNER, qW2, DINNER, DINNER, x2, nullptr, DMODEL, b2, x, gammaU);
  // 5) qx = groupwise quant(x2)
  quant_kernel<<<grids(NTOK * DMODEL), blk256, 0, stream>>>(x2, gammaU, qx);
  // 6) u = LN2(qx) -> bf16
  ln_bf16_kernel<<<dim3(NTOK), blk256, 0, stream>>>(qx, ln2_g, ln2_b, lnBf);
  // 7) xz = u @ in_proj^T
  gemm_bf16_kernel<2><<<dim3(NTOK / 64, 4096 / 64), blk128, 0, stream>>>(
      lnBf, DMODEL, wInBf, DMODEL, DMODEL, xzBuf, nullptr, 4096, nullptr, nullptr, nullptr);
  // 8) xs = silu(causal depthwise conv(xs) + conv_b)
  conv_silu_kernel<<<dim3(NTOK), blk256, 0, stream>>>(xzBuf, conv_w, conv_b, xscBf);
  // 9) x_dbl = xs @ x_proj^T (padded 96->128)
  gemm_bf16_kernel<3><<<dim3(NTOK / 64, 128 / 64), blk128, 0, stream>>>(
      xscBf, DINNER, wXpBf, DINNER, DINNER, xdbl, xdblBf, 128, nullptr, nullptr, nullptr);
  // 10) dt = softplus(x_dbl[:, :64] @ dt_proj^T + dt_proj_b)
  gemm_bf16_kernel<4><<<dim3(NTOK / 64, DINNER / 64), blk128, 0, stream>>>(
      xdblBf, 128, wDtBf, 64, 64, dtBuf, nullptr, DINNER, dt_projb, nullptr, nullptr);
  // 11) selective scan + D*x + silu(z) gate
  scan_kernel<<<dim3(DINNER / 256, BSZ), blk256, 0, stream>>>(
      dtBuf, xscBf, xzBuf, xdbl, Aneg, D_param, yBf);
  // 12) out = qx + y @ out_proj^T
  gemm_bf16_kernel<5><<<dim3(NTOK / 64, DMODEL / 64), blk128, 0, stream>>>(
      yBf, DINNER, wOutBf, DINNER, DINNER, out, nullptr, DMODEL, nullptr, qx, nullptr);
}
